// LPG_87101936763173
// MI455X (gfx1250) — matrix operand order repaired and checked by reading, not compiled, not run on hardware
//
#include <hip/hip_runtime.h>
#include <hip/hip_bf16.h>
#include <math.h>

// ---------------------------------------------------------------------------
// Sizes from the reference
// ---------------------------------------------------------------------------
#define B_  256
#define S_  512
#define T_  32
#define E_  32
#define H_  512
#define BS_ (B_*S_)

typedef __bf16 bf16;
typedef __attribute__((ext_vector_type(16))) __bf16 v16bf;
typedef __attribute__((ext_vector_type(8)))  float  v8f;

// One WMMA operand fragment: 16 bf16 per lane x 32 lanes (32B per lane)
union Frag { v16bf v; uint4 q[2]; unsigned u[8]; };

// ISA 16-bit A-matrix 16x32 swizzle: VGPR pair p (0..7), lane-half lh (0/1)
//   p<4 : K = 8*lh + 2p(+1)        p>=4 : K = 16 + 8*lh + 2(p-4)(+1)
__device__ __host__ __forceinline__ int kOffPair(int p, int lh) {
    return (p < 4) ? (8*lh + 2*p) : (16 + 8*lh + 2*(p-4));
}

// ---------------------------------------------------------------------------
// Kernel 1: per-(b,s) features [r, d, pi, mlp(yt), mask*mlp(yt1), step, life]
// stored bf16, PRE-PERMUTED so lane-half lh reads its 16 fragment values as
// one contiguous 32B run:  pos = lh*16 + j  holds  feature[koff(j,lh)].
// ---------------------------------------------------------------------------
__global__ void LPG_feat_kernel(const float* __restrict__ r, const int* __restrict__ d,
                                const float* __restrict__ pi,
                                const float* __restrict__ yt, const float* __restrict__ yt1,
                                const int* __restrict__ step, const int* __restrict__ life,
                                const float* __restrict__ W1, const float* __restrict__ b1,
                                const float* __restrict__ W2, const float* __restrict__ b2,
                                bf16* __restrict__ xfeat) {
    int idx = blockIdx.x * blockDim.x + threadIdx.x;
    if (idx >= BS_) return;
    const float* y0 = yt  + idx * T_;
    const float* y1 = yt1 + idx * T_;
    float ry0[T_], ry1[T_];
    for (int t = 0; t < T_; ++t) { ry0[t] = y0[t]; ry1[t] = y1[t]; }
    float p0 = b2[0], p1 = b2[0];
    for (int e = 0; e < E_; ++e) {
        float a0 = b1[e], a1 = b1[e];
        for (int t = 0; t < T_; ++t) {
            float w = W1[t * E_ + e];
            a0 += ry0[t] * w; a1 += ry1[t] * w;
        }
        float w2 = W2[e];
        p0 += fmaxf(a0, 0.f) * w2;
        p1 += fmaxf(a1, 0.f) * w2;
    }
    int dv = d[idx];
    float f[8];
    f[0] = r[idx];
    f[1] = dv ? 1.f : 0.f;
    f[2] = pi[idx];
    f[3] = p0;
    f[4] = dv ? 0.f : p1;       // pyt1 zeroed where done
    f[5] = (float)step[0];
    f[6] = (float)life[0];
    f[7] = 0.f;
    bf16* o = xfeat + idx * 32;
    #pragma unroll
    for (int pos = 0; pos < 32; ++pos) {
        int lh = pos >> 4, j = pos & 15;
        int ko = kOffPair(j >> 1, lh) + (j & 1);
        o[pos] = (bf16)((ko < 8) ? f[ko] : 0.f);
    }
}

// ---------------------------------------------------------------------------
// Kernel 2: repack weights into bf16 WMMA B-operand fragments, fragment-
// contiguous (32B per lane per fragment -> 2x global_load_b128).
//   whP : [gate(3)][ntile(32)][kchunk(16)] frags of Wh[k][n]   (r,z,n)
//   wiP : [ntile(96)] frags of Wi zero-padded to K=32
//   wypP: [nt3(3)][kchunk(16)] frags of [Wy | Wp | 0] (512x48)
// ---------------------------------------------------------------------------
#define WH_FRAGS  (3*32*16)
#define WI_FRAGS  96
#define WYP_FRAGS 48
#define WH_ELEMS  (WH_FRAGS*512)
#define WI_ELEMS  (WI_FRAGS*512)
#define WYP_ELEMS (WYP_FRAGS*512)

__global__ void LPG_pack_kernel(const float* __restrict__ Wi,
                                const float* __restrict__ Whr, const float* __restrict__ Whz,
                                const float* __restrict__ Whn,
                                const float* __restrict__ Wp,  const float* __restrict__ Wy,
                                bf16* __restrict__ whP, bf16* __restrict__ wiP,
                                bf16* __restrict__ wypP) {
    int idx = blockIdx.x * blockDim.x + threadIdx.x;
    const int total = WH_ELEMS + WI_ELEMS + WYP_ELEMS;
    if (idx >= total) return;
    if (idx < WH_ELEMS) {
        int frag = idx >> 9, within = idx & 511;
        int l = within >> 4, j = within & 15;
        int g = frag / 512, rem = frag - g * 512;
        int nt = rem >> 4, kc = rem & 15;
        int n  = nt * 16 + (l & 15);
        int ko = kOffPair(j >> 1, l >> 4) + (j & 1);
        int k  = kc * 32 + ko;
        const float* W = (g == 0) ? Whr : (g == 1) ? Whz : Whn;
        whP[idx] = (bf16)W[k * H_ + n];
    } else if (idx < WH_ELEMS + WI_ELEMS) {
        int e = idx - WH_ELEMS;
        int frag = e >> 9, within = e & 511;
        int l = within >> 4, j = within & 15;
        int n = frag * 16 + (l & 15);
        int k = kOffPair(j >> 1, l >> 4) + (j & 1);
        wiP[e] = (bf16)((k < 7) ? Wi[k * 3 * H_ + n] : 0.f);
    } else {
        int e = idx - WH_ELEMS - WI_ELEMS;
        int frag = e >> 9, within = e & 511;
        int l = within >> 4, j = within & 15;
        int nt3 = frag >> 4, kc = frag & 15;
        int n48 = nt3 * 16 + (l & 15);
        int k   = kc * 32 + kOffPair(j >> 1, l >> 4) + (j & 1);
        float v = (n48 < 32) ? Wy[k * T_ + n48] : (n48 == 32 ? Wp[k] : 0.f);
        wypP[e] = (bf16)v;
    }
}

// ---------------------------------------------------------------------------
// Kernel 3: persistent GRU. One workgroup per 16-row batch tile, 512 threads
// = 16 waves; wave w owns H-column tiles 2w and 2w+1. The bf16 hidden state
// lives in LDS in FRAGMENT-MAJOR layout [kc][lane][16] so A-operands are two
// ds_load_b128 per lane, no address math.
// ---------------------------------------------------------------------------
__global__ __launch_bounds__(512)
void LPG_gru_kernel(const bf16* __restrict__ xfeat, const int* __restrict__ d,
                    const bf16* __restrict__ whP, const bf16* __restrict__ wiP,
                    const bf16* __restrict__ wypP,
                    const float* __restrict__ bi, const float* __restrict__ bhn,
                    const float* __restrict__ by, const float* __restrict__ bp,
                    float* __restrict__ out) {
    __shared__ float h32[16 * H_];      // 32 KB fp32 hidden state, row-major
    __shared__ Frag  hfr[16 * 32];      // 16 KB bf16 hidden state, fragment-major
    __shared__ float stage[16 * 48];    //  3 KB head accumulation

    const int bt  = blockIdx.x;         // batch tile (16 rows)
    const int tid = threadIdx.x;
    const int w   = tid >> 5;           // wave 0..15
    const int l   = tid & 31;           // lane
    const int lh  = l >> 4;             // lane half
    const int ml  = l & 15;             // A-row / C-column lane index

    { // zero init h
        float4 z4 = {0.f,0.f,0.f,0.f};
        float4* h4 = (float4*)h32;
        for (int i = tid; i < 16*H_/4; i += 512) h4[i] = z4;
        uint4 zq = {0u,0u,0u,0u};
        hfr[tid].q[0] = zq; hfr[tid].q[1] = zq;
    }
    __syncthreads();

    const Frag* whF  = (const Frag*)whP;
    const Frag* wiF  = (const Frag*)wiP;
    const Frag* wypF = (const Frag*)wypP;
    const Frag* xF   = (const Frag*)xfeat;     // 2 frags (lane-halves) per (b,s)
    const float bp0  = bp[0];

    // Per-sub (two column tiles per wave) constants
    int   nn[2], sIdx[2];
    float bir[2], biz[2], bin[2], bhnn[2];
    #pragma unroll
    for (int sub = 0; sub < 2; ++sub) {
        int nt = 2*w + sub;
        int n  = nt*16 + ml;
        nn[sub]   = n;
        bir[sub]  = bi[n];
        biz[sub]  = bi[H_ + n];
        bin[sub]  = bi[2*H_ + n];
        bhnn[sub] = bhn[n];
        // inverse swizzle: where does h[m][n] live in fragment-major layout?
        int kcw = n >> 5, ko = n & 31;
        int lhw = (ko >> 3) & 1;
        int pw  = (ko < 16) ? ((ko & 7) >> 1) : (4 + ((ko & 7) >> 1));
        int jw  = 2*pw + (ko & 1);
        sIdx[sub] = (kcw*32 + 16*lhw)*16 + jw;  // + (row)*16 added per element
    }

    for (int t = S_ - 1; t >= 0; --t) {
        // -- phase0: reset done rows, zero head stage ------------------------
        {
            int row = tid >> 5;                       // 16 rows x 32 threads
            int rst = d[(bt*16 + row)*S_ + t];
            if (rst) {
                float4 z4 = {0.f,0.f,0.f,0.f};
                float4* h4 = (float4*)&h32[row*H_ + (l)*16];
                h4[0]=z4; h4[1]=z4; h4[2]=z4; h4[3]=z4;
            }
            // fragment-major: row m owns lanes {m, m+16} in every kc fragment
            int m2 = tid & 15, q = tid >> 4;          // q: kc(0..15) x half(0..1)
            int kc = q & 15, hb2 = q >> 4;
            int rst2 = d[(bt*16 + m2)*S_ + t];
            if (rst2) {
                uint4 zq = {0u,0u,0u,0u};
                Frag* fp = &hfr[kc*32 + m2 + 16*hb2];
                fp->q[0] = zq; fp->q[1] = zq;
            }
            if (tid < 16*48 - 512) {}                 // (stage zero below)
            for (int i = tid; i < 16*48; i += 512) stage[i] = 0.f;
        }
        __syncthreads();

        // -- phase1: WMMA accumulation --------------------------------------
        Frag xa;                                      // A = x_t (16x32)
        {
            int row = (bt*16 + ml)*S_ + t;
            xa = xF[row*2 + lh];                      // 2x global_load_b128
        }
        v8f zero8 = {};
        v8f acc_r[2], acc_z[2], acc_nx[2], acc_nh[2];
        #pragma unroll
        for (int sub = 0; sub < 2; ++sub) {
            int nt = 2*w + sub;
            acc_r[sub]  = __builtin_amdgcn_wmma_f32_16x16x32_bf16(false, xa.v, false, wiF[nt*32 + l].v,      (short)0, zero8, false, false);
            acc_z[sub]  = __builtin_amdgcn_wmma_f32_16x16x32_bf16(false, xa.v, false, wiF[(32 + nt)*32 + l].v, (short)0, zero8, false, false);
            acc_nx[sub] = __builtin_amdgcn_wmma_f32_16x16x32_bf16(false, xa.v, false, wiF[(64 + nt)*32 + l].v, (short)0, zero8, false, false);
            acc_nh[sub] = zero8;
        }
        #pragma unroll 2
        for (int kc = 0; kc < 16; ++kc) {
            Frag ha = hfr[kc*32 + l];                 // 2x ds_load_b128, no math
            #pragma unroll
            for (int sub = 0; sub < 2; ++sub) {
                int nt = 2*w + sub;
                acc_r[sub]  = __builtin_amdgcn_wmma_f32_16x16x32_bf16(false, ha.v, false, whF[((0*32+nt)*16+kc)*32 + l].v, (short)0, acc_r[sub],  false, false);
                acc_z[sub]  = __builtin_amdgcn_wmma_f32_16x16x32_bf16(false, ha.v, false, whF[((1*32+nt)*16+kc)*32 + l].v, (short)0, acc_z[sub],  false, false);
                acc_nh[sub] = __builtin_amdgcn_wmma_f32_16x16x32_bf16(false, ha.v, false, whF[((2*32+nt)*16+kc)*32 + l].v, (short)0, acc_nh[sub], false, false);
            }
        }
        __syncthreads();

        // -- phase2: gates (register-elementwise), write h_new --------------
        bf16* hbE = (bf16*)hfr;
        #pragma unroll
        for (int sub = 0; sub < 2; ++sub) {
            int n = nn[sub];
            #pragma unroll
            for (int v = 0; v < 8; ++v) {
                int   m    = lh*8 + v;                // C/D row for this VGPR
                float rg   = 1.f / (1.f + __expf(-(acc_r[sub][v] + bir[sub])));
                float zg   = 1.f / (1.f + __expf(-(acc_z[sub][v] + biz[sub])));
                float ng   = tanhf(acc_nx[sub][v] + bin[sub] + rg*(acc_nh[sub][v] + bhnn[sub]));
                float hp   = h32[m*H_ + n];
                float hnew = (1.f - zg)*ng + zg*hp;
                h32[m*H_ + n] = hnew;
                hbE[sIdx[sub] + m*16] = (bf16)hnew;   // constant-stride scatter
            }
        }
        __syncthreads();

        // -- phase3: head GEMM relu(h) @ [Wy|Wp|0] (48 WMMA jobs, 3/wave) ---
        #pragma unroll
        for (int jj = 0; jj < 3; ++jj) {
            int job = w + jj*16;
            int nt3 = job >> 4, kc = job & 15;
            Frag ya = hfr[kc*32 + l];
            #pragma unroll
            for (int p = 0; p < 8; ++p) {             // relu on packed bf16
                unsigned u = ya.u[p];
                if (u & 0x00008000u) u &= 0xFFFF0000u;
                if (u & 0x80000000u) u &= 0x0000FFFFu;
                ya.u[p] = u;
            }
            v8f acc = __builtin_amdgcn_wmma_f32_16x16x32_bf16(false, ya.v, false, wypF[job*32 + l].v, (short)0, zero8, false, false);
            int cn = nt3*16 + ml;
            #pragma unroll
            for (int v = 0; v < 8; ++v)
                atomicAdd(&stage[(lh*8 + v)*48 + cn], acc[v]);   // ds_add_f32
        }
        __syncthreads();

        // -- phase4: bias + softmax(T=32 within one wave) + store -----------
        {
            int m = tid >> 5, c = tid & 31;           // wave == row, lane == col
            int brow = bt*16 + m;
            float val = stage[m*48 + c] + by[c];
            if (c == 0) out[brow*S_ + t] = stage[m*48 + 32] + bp0;   // pi_hat
            float mx = val;
            for (int o = 16; o > 0; o >>= 1) mx = fmaxf(mx, __shfl_xor(mx, o, 32));
            float ev = __expf(val - mx);
            float sm = ev;
            for (int o = 16; o > 0; o >>= 1) sm += __shfl_xor(sm, o, 32);
            out[BS_ + (brow*S_ + t)*T_ + c] = ev / sm;               // y_hat
        }
        __syncthreads();
    }
}

// ---------------------------------------------------------------------------
extern "C" void kernel_launch(void* const* d_in, const int* in_sizes, int n_in,
                              void* d_out, int out_size, void* d_ws, size_t ws_size,
                              hipStream_t stream) {
    const float* r    = (const float*)d_in[0];
    const int*   d    = (const int*)  d_in[1];
    const float* pi   = (const float*)d_in[2];
    const float* yt   = (const float*)d_in[3];
    const float* yt1  = (const float*)d_in[4];
    const int*   step = (const int*)  d_in[5];
    const int*   life = (const int*)  d_in[6];
    const float* W1   = (const float*)d_in[7];
    const float* b1   = (const float*)d_in[8];
    const float* W2   = (const float*)d_in[9];
    const float* b2   = (const float*)d_in[10];
    const float* Wi   = (const float*)d_in[11];
    const float* bi   = (const float*)d_in[12];
    const float* Whr  = (const float*)d_in[13];
    const float* Whz  = (const float*)d_in[14];
    const float* Whn  = (const float*)d_in[15];
    const float* bhn  = (const float*)d_in[16];
    const float* Wp   = (const float*)d_in[17];
    const float* bp   = (const float*)d_in[18];
    const float* Wy   = (const float*)d_in[19];
    const float* by   = (const float*)d_in[20];

    char* ws = (char*)d_ws;
    bf16* xfeat = (bf16*)ws;                                        // 8 MB
    bf16* whP   = (bf16*)(ws + 8388608);                            // 1.5 MB
    bf16* wiP   = (bf16*)(ws + 8388608 + 1572864);                  // 96 KB
    bf16* wypP  = (bf16*)(ws + 8388608 + 1572864 + 98304);          // 48 KB
    float* out  = (float*)d_out;

    LPG_feat_kernel<<<(BS_ + 255) / 256, 256, 0, stream>>>(
        r, d, pi, yt, yt1, step, life, W1, b1, W2, b2, xfeat);
    const int packTotal = WH_ELEMS + WI_ELEMS + WYP_ELEMS;
    LPG_pack_kernel<<<(packTotal + 255) / 256, 256, 0, stream>>>(
        Wi, Whr, Whz, Whn, Wp, Wy, whP, wiP, wypP);
    LPG_gru_kernel<<<B_ / 16, 512, 0, stream>>>(
        xfeat, d, whP, wiP, wypP, bi, bhn, by, bp, out);
}
